// AttentionBlock_81767587381439
// MI455X (gfx1250) — matrix operand order repaired and checked by reading, not compiled, not run on hardware
//
#include <hip/hip_runtime.h>
#include <hip/hip_bf16.h>
#include <stdint.h>

#define C_DIM 512
#define N_TOK 4096
#define B_DIM 8
#define GROUPS 8
#define EPS 1e-5f

typedef __attribute__((ext_vector_type(16))) __bf16 v16bf;
typedef __attribute__((ext_vector_type(8)))  __bf16 v8bf;
typedef __attribute__((ext_vector_type(8)))  float  v8f;

// ---------------------------------------------------------------------------
// WMMA helper: D = A(16x32 bf16) * B(32x16 bf16) + C(16x16 f32)
// ---------------------------------------------------------------------------
__device__ __forceinline__ v8f wmma_bf16(v16bf a, v16bf b, v8f c) {
  return __builtin_amdgcn_wmma_f32_16x16x32_bf16(
      false, a, false, b, (short)0, c, false, false);
}

// A-fragment (16x32, row-major source, leading dim ld elements)
__device__ __forceinline__ v16bf load_a_frag(const __bf16* p, int ld, int row,
                                             int kbase, int lid) {
  int lr = lid & 15, hi = (lid >> 4) & 1;
  const __bf16* base = p + (size_t)(row + lr) * ld + kbase + hi * 8;
  v8bf lo = *(const v8bf*)(base);
  v8bf hv = *(const v8bf*)(base + 16);
  v16bf r;
#pragma unroll
  for (int i = 0; i < 8; ++i) { r[i] = lo[i]; r[8 + i] = hv[i]; }
  return r;
}

// B-fragment (32x16): element B[k][n] lives at p[n*ld + k]
__device__ __forceinline__ v16bf load_b_frag(const __bf16* p, int ld, int nbase,
                                             int kbase, int lid) {
  const __bf16* base =
      p + (size_t)(nbase + (lid & 15)) * ld + kbase + ((lid >> 4) << 3);
  v8bf lo = *(const v8bf*)(base);
  v8bf hv = *(const v8bf*)(base + 16);
  v16bf r;
#pragma unroll
  for (int i = 0; i < 8; ++i) { r[i] = lo[i]; r[8 + i] = hv[i]; }
  return r;
}

// ---------------------------------------------------------------------------
// gfx1250 async global->LDS copy (16B per lane) + counter waits (ASYNCcnt)
// ---------------------------------------------------------------------------
__device__ __forceinline__ void async_g2lds_b128(void* lds, const void* g) {
  uint32_t loff = (uint32_t)(uintptr_t)lds;  // low 32 bits of flat = LDS offset
  asm volatile("global_load_async_to_lds_b128 %0, %1, off"
               :: "v"(loff), "v"(g) : "memory");
}
__device__ __forceinline__ void wait_async0()  {
  asm volatile("s_wait_asynccnt 0" ::: "memory");
}
__device__ __forceinline__ void wait_async16() {  // newest 16 may stay in flight
  asm volatile("s_wait_asynccnt 16" ::: "memory");
}

// ---------------------------------------------------------------------------
// Kernel 0: convert weights f32 -> bf16
// ---------------------------------------------------------------------------
__global__ void k_cvtw(const float* wq, const float* wk, const float* wv,
                       const float* wo, __bf16* w4) {
  int i = blockIdx.x * blockDim.x + threadIdx.x;
  if (i < C_DIM * C_DIM) {
    w4[i]                     = (__bf16)wq[i];
    w4[C_DIM * C_DIM + i]     = (__bf16)wk[i];
    w4[2 * C_DIM * C_DIM + i] = (__bf16)wv[i];
    w4[3 * C_DIM * C_DIM + i] = (__bf16)wo[i];
  }
}

// ---------------------------------------------------------------------------
// Kernel 1: GroupNorm; x[b][c][n] f32 -> h[b][n][c] bf16
// ---------------------------------------------------------------------------
__global__ __launch_bounds__(1024) void k_gn(const float* __restrict__ x,
                                             const float* __restrict__ gw,
                                             const float* __restrict__ gb,
                                             __bf16* __restrict__ h) {
  __shared__ float rs[32], rs2[32];
  const int bg = blockIdx.x;
  const int b = bg >> 3, g = bg & 7;
  const int CH = C_DIM / GROUPS;
  const float* xp = x + ((size_t)b * C_DIM + g * CH) * N_TOK;
  const int total = CH * N_TOK;

  float s = 0.f, s2 = 0.f;
  for (int i = threadIdx.x; i < total; i += blockDim.x) {
    float v = xp[i];
    s += v; s2 += v * v;
  }
  int w = threadIdx.x >> 5, lid = threadIdx.x & 31;
#pragma unroll
  for (int off = 16; off >= 1; off >>= 1) {
    s  += __shfl_down(s, off);
    s2 += __shfl_down(s2, off);
  }
  if (lid == 0) { rs[w] = s; rs2[w] = s2; }
  __syncthreads();
  if (threadIdx.x == 0) {
    float S = 0.f, S2 = 0.f;
    for (int i = 0; i < 32; ++i) { S += rs[i]; S2 += rs2[i]; }
    rs[0] = S; rs2[0] = S2;
  }
  __syncthreads();
  const float inv_n = 1.0f / (float)total;
  const float mu  = rs[0] * inv_n;
  const float var = rs2[0] * inv_n - mu * mu;
  const float inv = rsqrtf(var + EPS);

  for (int i = threadIdx.x; i < total; i += blockDim.x) {
    int ci = i >> 12;
    int n  = i & (N_TOK - 1);
    int ch = g * CH + ci;
    float v = (xp[i] - mu) * inv * gw[ch] + gb[ch];
    h[((size_t)b * N_TOK + n) * C_DIM + ch] = (__bf16)v;
  }
}

// ---------------------------------------------------------------------------
// Kernel 2: fused QKV projection with async LDS-staged weight slice.
// ---------------------------------------------------------------------------
__global__ __launch_bounds__(256) void k_qkv(const __bf16* __restrict__ h,
                                             const __bf16* __restrict__ w4,
                                             const float* __restrict__ bq,
                                             const float* __restrict__ bk,
                                             const float* __restrict__ bv,
                                             __bf16* __restrict__ q,
                                             __bf16* __restrict__ k,
                                             __bf16* __restrict__ vt) {
  __shared__ __align__(16) __bf16 Wl[64 * C_DIM];  // 64 KB
  int bx = blockIdx.x;
  const int ng  = bx & 15; bx >>= 4;
  const int d64 = bx & 7;  bx >>= 3;
  const int mat = bx % 3;
  const int b   = bx / 3;
  const int lid = threadIdx.x & 31;
  const int w   = threadIdx.x >> 5;

  const __bf16* W = w4 + (size_t)mat * C_DIM * C_DIM + (size_t)(d64 * 64) * C_DIM;
  for (int i = threadIdx.x; i < 64 * C_DIM / 8; i += 256)
    async_g2lds_b128(&Wl[i * 8], W + (size_t)i * 8);
  wait_async0();
  __syncthreads();

  const __bf16* A = h + (size_t)b * N_TOK * C_DIM;
  const int row = ng * 256 + w * 32;
  const int dbase = d64 * 64;

  v8f acc[8] = {};
  v16bf a0 = load_a_frag(A, C_DIM, row, 0, lid);
  v16bf a1 = load_a_frag(A, C_DIM, row + 16, 0, lid);
  for (int ks = 0; ks < C_DIM; ks += 32) {
    v16bf na0, na1;
    if (ks + 32 < C_DIM) {
      na0 = load_a_frag(A, C_DIM, row, ks + 32, lid);
      na1 = load_a_frag(A, C_DIM, row + 16, ks + 32, lid);
    }
#pragma unroll
    for (int nt = 0; nt < 4; ++nt) {
      v16bf bf = load_b_frag(Wl, C_DIM, nt * 16, ks, lid);
      acc[nt]     = wmma_bf16(a0, bf, acc[nt]);
      acc[4 + nt] = wmma_bf16(a1, bf, acc[4 + nt]);
    }
    a0 = na0; a1 = na1;
  }

  const float* bias = (mat == 0) ? bq : (mat == 1) ? bk : bv;
  const int hi = lid >> 4, col0 = lid & 15;
  if (mat < 2) {
    __bf16* out = ((mat == 0) ? q : k) + (size_t)b * N_TOK * C_DIM;
#pragma unroll
    for (int rg = 0; rg < 2; ++rg)
#pragma unroll
      for (int nt = 0; nt < 4; ++nt) {
        int d = dbase + nt * 16 + col0;
        float bi = bias[d];
#pragma unroll
        for (int r = 0; r < 8; ++r) {
          int rr = row + rg * 16 + r + hi * 8;
          out[(size_t)rr * C_DIM + d] = (__bf16)(acc[rg * 4 + nt][r] + bi);
        }
      }
  } else {
    __bf16* out = vt + (size_t)b * C_DIM * N_TOK;
#pragma unroll
    for (int rg = 0; rg < 2; ++rg)
#pragma unroll
      for (int nt = 0; nt < 4; ++nt) {
        int d = dbase + nt * 16 + col0;
        float bi = bias[d];
#pragma unroll
        for (int r = 0; r < 8; ++r) {
          int rr = row + rg * 16 + r + hi * 8;
          out[(size_t)d * N_TOK + rr] = (__bf16)(acc[rg * 4 + nt][r] + bi);
        }
      }
  }
}

// ---------------------------------------------------------------------------
// Kernel 3: flash attention with double-buffered async K/V staging.
// block = 512 thr = 16 waves, 64 query rows; wave (mi 0..3, di 0..3).
// LDS: Kl[2][64][520] + Vl[2][512][72] (padded: lane stride = 4 dwords mod 64
// -> conflict-free ds_load_b128) + S[64][68] + Pbuf[64][72] + states.
// Pipeline: issue async stage of block kb+1, s_wait_asynccnt 16 to guarantee
// block kb resident while kb+1 stays in flight.
// ---------------------------------------------------------------------------
#define BM 64
#define BN 64
#define KLD 520
#define VLD 72
#define SPAD 68
#define PPAD 72

__device__ __forceinline__ void stage_kv(__bf16* Kl, __bf16* Vl,
                                         const __bf16* K, const __bf16* VT,
                                         int kb, int tid) {
  // K block: 64 rows x 512 d -> 4096 x 16B chunks
#pragma unroll
  for (int j = 0; j < 8; ++j) {
    int item = tid + j * 512;
    int rowk = item >> 6;
    int ck   = (item & 63) * 8;
    async_g2lds_b128(Kl + rowk * KLD + ck, K + (size_t)(kb + rowk) * C_DIM + ck);
  }
  // V slice: 512 rows(d) x 64 m -> 4096 x 16B chunks
#pragma unroll
  for (int j = 0; j < 8; ++j) {
    int item = tid + j * 512;
    int rowd = item >> 3;
    int cm   = (item & 7) * 8;
    async_g2lds_b128(Vl + rowd * VLD + cm, VT + (size_t)rowd * N_TOK + kb + cm);
  }
}

__global__ __launch_bounds__(512) void k_flash(const __bf16* __restrict__ q,
                                               const __bf16* __restrict__ kk,
                                               const __bf16* __restrict__ vt,
                                               __bf16* __restrict__ o) {
  __shared__ __align__(16) __bf16 Kl[2][BM * KLD];   // 130 KB
  __shared__ __align__(16) __bf16 Vl[2][C_DIM * VLD];// 144 KB
  __shared__ float S[BM][SPAD];                      // 17.4 KB
  __shared__ __align__(16) __bf16 Pbuf[BM][PPAD];    // 9.2 KB
  __shared__ float mstate[BM], lstate[BM], alpha[BM];

  const int b  = blockIdx.x >> 6;
  const int qb = (blockIdx.x & 63) * BM;
  const int w = threadIdx.x >> 5, lid = threadIdx.x & 31;
  const int mi = w & 3, di = w >> 2;
  const int hi = lid >> 4, col0 = lid & 15;

  const __bf16* Q  = q  + (size_t)b * N_TOK * C_DIM;
  const __bf16* K  = kk + (size_t)b * N_TOK * C_DIM;
  const __bf16* VT = vt + (size_t)b * C_DIM * N_TOK;

  // kick off staging of the first K/V block, overlap with Q fragment loads
  stage_kv(Kl[0], Vl[0], K, VT, 0, threadIdx.x);

  v16bf qf[4];
#pragma unroll
  for (int i = 0; i < 4; ++i)
    qf[i] = load_a_frag(Q, C_DIM, qb + mi * 16, di * 128 + i * 32, lid);

  v8f oacc[8] = {};
  if (threadIdx.x < BM) { mstate[threadIdx.x] = -1e30f; lstate[threadIdx.x] = 0.f; }

  const float scale = 0.044194173824159216f;  // 1/sqrt(512)

  for (int kb = 0; kb < N_TOK; kb += BN) {
    const int cur = (kb >> 6) & 1;
    // stage next block into the other buffer; wait for current block only
    if (kb + BN < N_TOK) {
      stage_kv(Kl[cur ^ 1], Vl[cur ^ 1], K, VT, kb + BN, threadIdx.x);
      wait_async16();
    } else {
      wait_async0();
    }
    // zero shared score tile (consumed via ds_add_f32)
    for (int i = threadIdx.x; i < BM * SPAD; i += 512) ((float*)S)[i] = 0.f;
    __syncthreads();  // cur buffer resident everywhere, S zeroed

    // partial scores over this wave's 128-wide d-slice (B frags from LDS)
    v8f sacc[4] = {};
#pragma unroll
    for (int ks = 0; ks < 4; ++ks) {
#pragma unroll
      for (int nt = 0; nt < 4; ++nt) {
        v16bf bf = load_b_frag(Kl[cur], KLD, nt * 16, di * 128 + ks * 32, lid);
        sacc[nt] = wmma_bf16(qf[ks], bf, sacc[nt]);
      }
    }
    // split-K reduction across di waves: ds_add_f32 into S
#pragma unroll
    for (int nt = 0; nt < 4; ++nt)
#pragma unroll
      for (int r = 0; r < 8; ++r)
        atomicAdd(&S[mi * 16 + r + hi * 8][nt * 16 + col0], sacc[nt][r]);
    __syncthreads();

    // online softmax: thread t -> row t>>3, 8 cols
    {
      int row = threadIdx.x >> 3;
      int c0  = (threadIdx.x & 7) * 8;
      float sv[8];
#pragma unroll
      for (int j = 0; j < 8; ++j) sv[j] = S[row][c0 + j] * scale;
      float mx = sv[0];
#pragma unroll
      for (int j = 1; j < 8; ++j) mx = fmaxf(mx, sv[j]);
#pragma unroll
      for (int off = 4; off >= 1; off >>= 1) mx = fmaxf(mx, __shfl_xor(mx, off, 8));
      float mold = mstate[row];
      float mnew = fmaxf(mold, mx);
      float psum = 0.f;
#pragma unroll
      for (int j = 0; j < 8; ++j) {
        float p = __expf(sv[j] - mnew);
        psum += p;
        Pbuf[row][c0 + j] = (__bf16)p;
      }
#pragma unroll
      for (int off = 4; off >= 1; off >>= 1) psum += __shfl_xor(psum, off, 8);
      if ((threadIdx.x & 7) == 0) {
        float al = __expf(mold - mnew);
        alpha[row]  = al;
        mstate[row] = mnew;
        lstate[row] = lstate[row] * al + psum;
      }
    }
    __syncthreads();

    // rescale O, then O += P * V (B frags from staged Vl)
    {
      float al[8];
#pragma unroll
      for (int r = 0; r < 8; ++r) al[r] = alpha[mi * 16 + r + hi * 8];
#pragma unroll
      for (int t = 0; t < 8; ++t)
#pragma unroll
        for (int r = 0; r < 8; ++r) oacc[t][r] *= al[r];

      v16bf pf[2];
#pragma unroll
      for (int mk = 0; mk < 2; ++mk) {
        int lr = lid & 15;
        const __bf16* basep = &Pbuf[mi * 16 + lr][mk * 32 + hi * 8];
        v8bf lo = *(const v8bf*)basep;
        v8bf hv = *(const v8bf*)(basep + 16);
#pragma unroll
        for (int i = 0; i < 8; ++i) { pf[mk][i] = lo[i]; pf[mk][8 + i] = hv[i]; }
      }
#pragma unroll
      for (int nt = 0; nt < 8; ++nt) {
#pragma unroll
        for (int mk = 0; mk < 2; ++mk) {
          v16bf bf = load_b_frag(Vl[cur], VLD, di * 128 + nt * 16, mk * 32, lid);
          oacc[nt] = wmma_bf16(pf[mk], bf, oacc[nt]);
        }
      }
    }
    __syncthreads();  // all reads of cur done before it is restaged
  }

  // finalize: O /= l, write bf16 [b][n][d]
  float li[8];
#pragma unroll
  for (int r = 0; r < 8; ++r) li[r] = 1.0f / lstate[mi * 16 + r + hi * 8];
  __bf16* O = o + (size_t)b * N_TOK * C_DIM;
#pragma unroll
  for (int nt = 0; nt < 8; ++nt) {
    int d = di * 128 + nt * 16 + col0;
#pragma unroll
    for (int r = 0; r < 8; ++r) {
      int rr = qb + mi * 16 + r + hi * 8;
      O[(size_t)rr * C_DIM + d] = (__bf16)(oacc[nt][r] * li[r]);
    }
  }
}

// ---------------------------------------------------------------------------
// Kernel 4: output projection + bias + residual -> d_out f32 [B][C][N]
// ---------------------------------------------------------------------------
__global__ __launch_bounds__(256) void k_oproj(const __bf16* __restrict__ ao,
                                               const __bf16* __restrict__ wob,
                                               const float* __restrict__ bo,
                                               const float* __restrict__ x,
                                               float* __restrict__ out) {
  __shared__ __align__(16) __bf16 Wl[64 * C_DIM];  // 64 KB
  int bx = blockIdx.x;
  const int ng  = bx & 15; bx >>= 4;
  const int c64 = bx & 7;  bx >>= 3;
  const int b   = bx;
  const int lid = threadIdx.x & 31;
  const int w   = threadIdx.x >> 5;

  const __bf16* W = wob + (size_t)(c64 * 64) * C_DIM;
  for (int i = threadIdx.x; i < 64 * C_DIM / 8; i += 256)
    async_g2lds_b128(&Wl[i * 8], W + (size_t)i * 8);
  wait_async0();
  __syncthreads();

  const __bf16* A = ao + (size_t)b * N_TOK * C_DIM;
  const int row = ng * 256 + w * 32;
  const int cbase = c64 * 64;

  v8f acc[8] = {};
  v16bf a0 = load_a_frag(A, C_DIM, row, 0, lid);
  v16bf a1 = load_a_frag(A, C_DIM, row + 16, 0, lid);
  for (int ks = 0; ks < C_DIM; ks += 32) {
    v16bf na0, na1;
    if (ks + 32 < C_DIM) {
      na0 = load_a_frag(A, C_DIM, row, ks + 32, lid);
      na1 = load_a_frag(A, C_DIM, row + 16, ks + 32, lid);
    }
#pragma unroll
    for (int nt = 0; nt < 4; ++nt) {
      v16bf bf = load_b_frag(Wl, C_DIM, nt * 16, ks, lid);
      acc[nt]     = wmma_bf16(a0, bf, acc[nt]);
      acc[4 + nt] = wmma_bf16(a1, bf, acc[4 + nt]);
    }
    a0 = na0; a1 = na1;
  }

  const int hi = lid >> 4, col0 = lid & 15;
#pragma unroll
  for (int rg = 0; rg < 2; ++rg)
#pragma unroll
    for (int nt = 0; nt < 4; ++nt) {
      int c = cbase + nt * 16 + col0;
      float bi = bo[c];
      const float* xp = x   + ((size_t)b * C_DIM + c) * N_TOK + row + rg * 16 + hi * 8;
      float*       op = out + ((size_t)b * C_DIM + c) * N_TOK + row + rg * 16 + hi * 8;
#pragma unroll
      for (int r = 0; r < 8; ++r) op[r] = xp[r] + acc[rg * 4 + nt][r] + bi;
    }
}

// ---------------------------------------------------------------------------
// host launcher
// ---------------------------------------------------------------------------
extern "C" void kernel_launch(void* const* d_in, const int* in_sizes, int n_in,
                              void* d_out, int out_size, void* d_ws, size_t ws_size,
                              hipStream_t stream) {
  (void)in_sizes; (void)n_in; (void)out_size; (void)ws_size;
  const float* x    = (const float*)d_in[0];
  const float* gn_w = (const float*)d_in[1];
  const float* gn_b = (const float*)d_in[2];
  const float* wq   = (const float*)d_in[3];
  const float* bq   = (const float*)d_in[4];
  const float* wk   = (const float*)d_in[5];
  const float* bk   = (const float*)d_in[6];
  const float* wv   = (const float*)d_in[7];
  const float* bv   = (const float*)d_in[8];
  const float* wo   = (const float*)d_in[9];
  const float* bo   = (const float*)d_in[10];
  float* out = (float*)d_out;

  const size_t elems = (size_t)B_DIM * N_TOK * C_DIM;
  char* ws = (char*)d_ws;
  __bf16* hb  = (__bf16*)(ws);
  __bf16* qb  = (__bf16*)(ws + elems * 2);
  __bf16* kb  = (__bf16*)(ws + elems * 4);
  __bf16* vtb = (__bf16*)(ws + elems * 6);
  __bf16* ob  = (__bf16*)(ws + elems * 8);
  __bf16* w4  = (__bf16*)(ws + elems * 10);

  k_cvtw<<<(C_DIM * C_DIM + 255) / 256, 256, 0, stream>>>(wq, wk, wv, wo, w4);
  k_gn<<<B_DIM * GROUPS, 1024, 0, stream>>>(x, gn_w, gn_b, hb);
  k_qkv<<<16 * 8 * 3 * B_DIM, 256, 0, stream>>>(hb, w4, bq, bk, bv, qb, kb, vtb);
  k_flash<<<B_DIM * (N_TOK / BM), 512, 0, stream>>>(qb, kb, vtb, ob);
  k_oproj<<<16 * 8 * B_DIM, 256, 0, stream>>>(
      ob, w4 + (size_t)3 * C_DIM * C_DIM, bo, x, out);
}
